// SDPAFlex_46797963657670
// MI455X (gfx1250) — compile-verified, matrix operand order checked by reading, not run on hardware
//
#include <hip/hip_runtime.h>

typedef __attribute__((ext_vector_type(16))) __bf16 v16bf;
typedef __attribute__((ext_vector_type(8)))  __bf16 v8bf;
typedef __attribute__((ext_vector_type(8)))  float  v8f;
typedef __attribute__((ext_vector_type(4)))  unsigned tdm_u4;
typedef __attribute__((ext_vector_type(8)))  int      tdm_i8;
typedef __attribute__((ext_vector_type(4)))  int      tdm_i4;

#define SEQ    2048
#define HD     128
#define NH     32
#define NKV    8
#define QTILE  128   // q rows per workgroup (16 per wave x 8 waves)
#define KTILE  32    // keys per inner block
#define KPITCH 136   // 128 + 8 bf16 pad (272B = 64+4 dwords, via TDM pad)
#define VPITCH 40    // 32 + 8 bf16 pad  (80B = 16+4 dwords, via TDM pad)
#define PPITCH 40

static __device__ __forceinline__ __bf16 f2bf(float f) {
  unsigned u = __builtin_bit_cast(unsigned, f);
  unsigned r = (u + 0x7FFFu + ((u >> 16) & 1u)) >> 16;  // round-to-nearest-even
  return __builtin_bit_cast(__bf16, (unsigned short)r);
}

// ---- TDM: issue a 2D tensor tile load (bf16) from global into LDS with row padding.
static __device__ __forceinline__ void tdm_load_2d(unsigned lds_off, const void* gaddr,
                                                   unsigned g1w0,
                                                   unsigned td0, unsigned td1,
                                                   unsigned tile0, unsigned tile1,
                                                   unsigned stride0) {
  unsigned long long ga = (unsigned long long)(size_t)gaddr;
  tdm_u4 g0;
  g0[0] = 1u;                                   // count=1, user descriptor
  g0[1] = lds_off;                              // D#.lds_addr
  g0[2] = (unsigned)(ga & 0xFFFFFFFFu);         // global_addr[31:0]
  g0[3] = (unsigned)((ga >> 32) & 0x01FFFFFFu)  // global_addr[56:32]
          | 0x80000000u;                        // type = 2 ("image")
  tdm_i8 g1;
  g1[0] = (int)g1w0;                                            // mask=0 | data_size | pad cfg
  g1[1] = (int)((td0 & 0xFFFFu) << 16);                         // barrier_addr=0 | tensor_dim0.lo
  g1[2] = (int)(((td0 >> 16) & 0xFFFFu) | ((td1 & 0xFFFFu) << 16));
  g1[3] = (int)(((td1 >> 16) & 0xFFFFu) | ((tile0 & 0xFFFFu) << 16));
  g1[4] = (int)(tile1 & 0xFFFFu);                               // tile_dim2 = 0
  g1[5] = (int)stride0;                                         // tensor_dim0_stride[31:0]
  g1[6] = 0;                                                    // stride0.hi | stride1.lo
  g1[7] = 0;
  tdm_i4 z4 = {0, 0, 0, 0};
#if defined(__clang_major__) && (__clang_major__ >= 23)
  tdm_i8 z8 = {0, 0, 0, 0, 0, 0, 0, 0};
  __builtin_amdgcn_tensor_load_to_lds(g0, g1, z4, z4, z8, 0);
#else
  __builtin_amdgcn_tensor_load_to_lds(g0, g1, z4, z4, 0);
#endif
}

// K tile: rows of 128 bf16 = 64 dwords, pad 4 dwords -> pitch 272B. data_size=2B.
#define G1W0_K ((1u << 16) | (1u << 20) | (5u << 22) | (3u << 25))
// V tile: rows of 32 bf16 = 16 dwords, pad 4 dwords -> pitch 80B.
#define G1W0_V ((1u << 16) | (1u << 20) | (3u << 22) | (3u << 25))

// ---- Pre-pass: convert K -> bf16 (same layout), V -> bf16 transposed [h][d][s].
__global__ __launch_bounds__(256)
void cvt_kv(const float* __restrict__ kg, const float* __restrict__ vg,
            __bf16* __restrict__ kbf, __bf16* __restrict__ vt) {
  const size_t idx = ((size_t)blockIdx.x * 256 + threadIdx.x) * 4;
  if (idx >= (size_t)NKV * SEQ * HD) return;
  float4 kk = *(const float4*)(kg + idx);
  kbf[idx + 0] = f2bf(kk.x); kbf[idx + 1] = f2bf(kk.y);
  kbf[idx + 2] = f2bf(kk.z); kbf[idx + 3] = f2bf(kk.w);
  const int d = (int)(idx % HD);
  const size_t t = idx / HD;
  const int s = (int)(t % SEQ);
  const int hh = (int)(t / SEQ);
  float4 vv = *(const float4*)(vg + idx);
  __bf16* vtp = vt + ((size_t)hh * HD + d) * SEQ + s;
  vtp[0 * SEQ] = f2bf(vv.x); vtp[1 * SEQ] = f2bf(vv.y);
  vtp[2 * SEQ] = f2bf(vv.z); vtp[3 * SEQ] = f2bf(vv.w);
}

// ---- Online-softmax step in C-layout (row = i + 8*half, col = n / 16+n).
// Row sums kept as per-lane partials (corr uses the reduced max); reduced in epilogue.
template <bool MASKED>
static __device__ __forceinline__ void softmax_step(v8f& S0, v8f& S1, v8f O[8],
                                                    float run_max[8], float run_sum[8],
                                                    __bf16 (*sPw)[PPITCH],
                                                    int kstart, int qbase, int n, int half) {
#pragma unroll
  for (int i = 0; i < 8; ++i) {
    float s0 = S0[i], s1 = S1[i];
    if (MASKED) {
      const int qrow = qbase + i + 8 * half;
      if (kstart + n > qrow)      s0 = -1e30f;
      if (kstart + 16 + n > qrow) s1 = -1e30f;
    }
    float mx = fmaxf(s0, s1);
#pragma unroll
    for (int m = 1; m < 16; m <<= 1) mx = fmaxf(mx, __shfl_xor(mx, m, 32));
    const float nm   = fmaxf(run_max[i], mx);
    const float corr = __expf(run_max[i] - nm);
    run_max[i] = nm;
    const float p0 = __expf(s0 - nm);
    const float p1 = __expf(s1 - nm);
    run_sum[i] = run_sum[i] * corr + (p0 + p1);
#pragma unroll
    for (int d = 0; d < 8; ++d) O[d][i] *= corr;
    sPw[i + 8 * half][n]      = f2bf(p0);
    sPw[i + 8 * half][16 + n] = f2bf(p1);
  }
}

__global__ __launch_bounds__(256, 1)
void sdpa_flash_wmma(const float* __restrict__ qg,
                     const __bf16* __restrict__ kbf,
                     const __bf16* __restrict__ vt,
                     float* __restrict__ out) {
  // Ping-pong TDM-staged tiles (double buffered) + per-wave P staging.
  __shared__ __align__(16) __bf16 sK[2][KTILE][KPITCH];
  __shared__ __align__(16) __bf16 sVt[2][HD][VPITCH];
  __shared__ __align__(16) __bf16 sP[8][16][PPITCH];

  const int tid  = threadIdx.x;
  const int lane = tid & 31;
  const int wave = tid >> 5;
  const int n    = lane & 15;
  const int half = lane >> 4;

  const int h     = blockIdx.y;
  const int kvh   = h >> 2;                 // GQA: 4 q-heads per kv-head
  const int q0    = blockIdx.x * QTILE;
  const int qbase = q0 + wave * 16;
  const int qlast = qbase + 15;

  const unsigned ldsK0 = (unsigned)(size_t)&sK[0][0][0];
  const unsigned ldsK1 = (unsigned)(size_t)&sK[1][0][0];
  const unsigned ldsV0 = (unsigned)(size_t)&sVt[0][0][0];
  const unsigned ldsV1 = (unsigned)(size_t)&sVt[1][0][0];
  const __bf16* kHead = kbf + (size_t)kvh * SEQ * HD;        // [seq][dim]
  const __bf16* vHead = vt + (size_t)kvh * HD * SEQ;         // [dim][seq]

  // ---- Q stripe (16 x 128) as four 16x32 bf16 A-fragments, 1/sqrt(d) pre-folded.
  const float scale = 0.08838834764831845f;
  v16bf A[4];
  {
    const float* qp = qg + ((size_t)h * SEQ + (qbase + n)) * HD;
#pragma unroll
    for (int f = 0; f < 4; ++f) {
      const int k0 = 32 * f + 8 * half;
      float4 a0 = *(const float4*)(qp + k0);
      float4 a1 = *(const float4*)(qp + k0 + 4);
      float4 b0 = *(const float4*)(qp + k0 + 16);
      float4 b1 = *(const float4*)(qp + k0 + 20);
      v16bf t;
      t[0]  = f2bf(a0.x * scale); t[1]  = f2bf(a0.y * scale);
      t[2]  = f2bf(a0.z * scale); t[3]  = f2bf(a0.w * scale);
      t[4]  = f2bf(a1.x * scale); t[5]  = f2bf(a1.y * scale);
      t[6]  = f2bf(a1.z * scale); t[7]  = f2bf(a1.w * scale);
      t[8]  = f2bf(b0.x * scale); t[9]  = f2bf(b0.y * scale);
      t[10] = f2bf(b0.z * scale); t[11] = f2bf(b0.w * scale);
      t[12] = f2bf(b1.x * scale); t[13] = f2bf(b1.y * scale);
      t[14] = f2bf(b1.z * scale); t[15] = f2bf(b1.w * scale);
      A[f] = t;
    }
  }

  const v8f vzero = {0.f, 0.f, 0.f, 0.f, 0.f, 0.f, 0.f, 0.f};
  v8f O[8];
#pragma unroll
  for (int d = 0; d < 8; ++d) O[d] = vzero;
  float run_max[8], run_sum[8];
#pragma unroll
  for (int i = 0; i < 8; ++i) { run_max[i] = -1e30f; run_sum[i] = 0.f; }

  const int nb = (q0 + QTILE) / KTILE;

  // Prologue: stage tile 0 into buffer 0 (DMA runs while Q fragments finish).
  if (wave == 0) {
    tdm_load_2d(ldsK0, kHead, G1W0_K, HD, SEQ, HD, KTILE, HD);
    tdm_load_2d(ldsV0, vHead, G1W0_V, SEQ, HD, KTILE, HD, SEQ);
  }

  for (int blk = 0; blk < nb; ++blk) {
    const int kstart = blk * KTILE;
    const int cur = blk & 1;

    __syncthreads();   // all waves done reading buf[cur^1] (tile blk-1): safe to overwrite
    if (wave == 0) {
      if (blk + 1 < nb) {
        const int knext = kstart + KTILE;
        tdm_load_2d(cur ? ldsK0 : ldsK1, kHead + (size_t)knext * HD, G1W0_K,
                    HD, SEQ, HD, KTILE, HD);
        tdm_load_2d(cur ? ldsV0 : ldsV1, vHead + knext, G1W0_V,
                    SEQ, HD, KTILE, HD, SEQ);
        // In-order TDM: leave the 2 just-issued in flight, require pair `blk` done.
        __builtin_amdgcn_s_wait_tensorcnt(2);
      } else {
        __builtin_amdgcn_s_wait_tensorcnt(0);
      }
    }
    __syncthreads();   // tile blk visible to all waves; tile blk+1 DMA overlaps compute

    if (kstart > qlast) continue;          // wave-uniform: EXEC all-ones below

    // ---- S = Q * K^T for this 32-key block.
    v8f S0 = vzero, S1 = vzero;
#pragma unroll
    for (int f = 0; f < 4; ++f) {
      const int k0 = 32 * f + 8 * half;
      v8bf lo0 = *reinterpret_cast<const v8bf*>(&sK[cur][n][k0]);
      v8bf hi0 = *reinterpret_cast<const v8bf*>(&sK[cur][n][k0 + 16]);
      v8bf lo1 = *reinterpret_cast<const v8bf*>(&sK[cur][16 + n][k0]);
      v8bf hi1 = *reinterpret_cast<const v8bf*>(&sK[cur][16 + n][k0 + 16]);
      v16bf B0 = __builtin_shufflevector(lo0, hi0, 0,1,2,3,4,5,6,7,8,9,10,11,12,13,14,15);
      v16bf B1 = __builtin_shufflevector(lo1, hi1, 0,1,2,3,4,5,6,7,8,9,10,11,12,13,14,15);
      S0 = __builtin_amdgcn_wmma_f32_16x16x32_bf16(false, A[f], false, B0, (short)0, S0, false, false);
      S1 = __builtin_amdgcn_wmma_f32_16x16x32_bf16(false, A[f], false, B1, (short)0, S1, false, false);
    }

    // ---- Online softmax; masked path only on diagonal blocks (wave-uniform test).
    if (kstart + KTILE - 1 <= qbase)
      softmax_step<false>(S0, S1, O, run_max, run_sum, sP[wave], kstart, qbase, n, half);
    else
      softmax_step<true>(S0, S1, O, run_max, run_sum, sP[wave], kstart, qbase, n, half);
    asm volatile("s_wait_dscnt 0" ::: "memory");   // same-wave LDS write->read

    // ---- O += P * V.
    v8bf plo = *reinterpret_cast<const v8bf*>(&sP[wave][n][8 * half]);
    v8bf phi = *reinterpret_cast<const v8bf*>(&sP[wave][n][16 + 8 * half]);
    v16bf P = __builtin_shufflevector(plo, phi, 0,1,2,3,4,5,6,7,8,9,10,11,12,13,14,15);
#pragma unroll
    for (int d = 0; d < 8; ++d) {
      v8bf vlo = *reinterpret_cast<const v8bf*>(&sVt[cur][16 * d + n][8 * half]);
      v8bf vhi = *reinterpret_cast<const v8bf*>(&sVt[cur][16 * d + n][16 + 8 * half]);
      v16bf Bv = __builtin_shufflevector(vlo, vhi, 0,1,2,3,4,5,6,7,8,9,10,11,12,13,14,15);
      O[d] = __builtin_amdgcn_wmma_f32_16x16x32_bf16(false, P, false, Bv, (short)0, O[d], false, false);
    }
  }

  // ---- Epilogue: reduce row sums (deferred), normalize, store fp32.
#pragma unroll
  for (int i = 0; i < 8; ++i) {
    float rs = run_sum[i];
#pragma unroll
    for (int m = 1; m < 16; m <<= 1) rs += __shfl_xor(rs, m, 32);
    const float inv = 1.0f / rs;
    const int qrow = qbase + i + 8 * half;
    float* op = out + (size_t)qrow * (NH * HD) + h * HD;
#pragma unroll
    for (int d = 0; d < 8; ++d) op[16 * d + n] = O[d][i] * inv;
  }
}

extern "C" void kernel_launch(void* const* d_in, const int* in_sizes, int n_in,
                              void* d_out, int out_size, void* d_ws, size_t ws_size,
                              hipStream_t stream) {
  (void)in_sizes; (void)n_in; (void)out_size; (void)ws_size;
  // setup_inputs order: input_pos, q, k, v, bsz, seqlen, mask
  const float* q = (const float*)d_in[1];
  const float* k = (const float*)d_in[2];
  const float* v = (const float*)d_in[3];
  float* out = (float*)d_out;

  __bf16* kbf = (__bf16*)d_ws;                                       // 4 MB
  __bf16* vt  = (__bf16*)((char*)d_ws + (size_t)NKV * SEQ * HD * 2); // 4 MB

  const int total4 = NKV * SEQ * HD / 4;
  cvt_kv<<<dim3((total4 + 255) / 256), dim3(256), 0, stream>>>(k, v, kbf, vt);

  dim3 grid(SEQ / QTILE, NH);   // (16 q-tiles, 32 heads)
  sdpa_flash_wmma<<<grid, dim3(256), 0, stream>>>(q, kbf, vt, out);
}
